// MultiHeadAttention_57612691308547
// MI455X (gfx1250) — matrix operand order repaired and checked by reading, not compiled, not run on hardware
//
#include <hip/hip_runtime.h>
#include <hip/hip_bf16.h>

// ---------------------------------------------------------------------------
// MI455X (gfx1250) multi-head attention, bf16 WMMA pipeline + TDM tile staging.
//   B=2, S=2048, E=1024, H=16, D=64, causal.
// ---------------------------------------------------------------------------

typedef __bf16 bf16_t;
typedef __attribute__((ext_vector_type(16))) __bf16 v16bf;
typedef __attribute__((ext_vector_type(8)))  __bf16 v8bf;
typedef __attribute__((ext_vector_type(4)))  __bf16 v4bf;
typedef __attribute__((ext_vector_type(8)))  float  v8f;
typedef __attribute__((ext_vector_type(4)))  unsigned int v4u;
typedef __attribute__((ext_vector_type(8)))  int v8i;
typedef __attribute__((ext_vector_type(4)))  int v4i;

#define WMMA_BF16(a, b, c) \
  __builtin_amdgcn_wmma_f32_16x16x32_bf16(false, (a), false, (b), (short)0, (c), false, false)

#if defined(__has_builtin)
#if __has_builtin(__builtin_amdgcn_tensor_load_to_lds) && \
    __has_builtin(__builtin_amdgcn_s_wait_tensorcnt)
#define USE_TDM 1
#endif
#endif
#ifndef USE_TDM
#define USE_TDM 0
#endif

static __device__ __forceinline__ v16bf combine16(v8bf lo, v8bf hi) {
  v16bf r;
#pragma unroll
  for (int e = 0; e < 8; ++e) { r[e] = lo[e]; r[e + 8] = hi[e]; }
  return r;
}

#if USE_TDM
// ---------------------------------------------------------------------------
// Issue one TDM 2D tile load: tile_d1 rows x tile_d0 elements (bf16) from a
// row-major tensor with row stride `stride0` (elements) into LDS at lds_addr.
// LDS rows are padded +8 bf16 per 64 bf16 (pad_interval=32 DW, pad_amount=4 DW).
// This toolchain's builtin takes 6 args: (v4u g0, v8i g1, v4i g2, v4i g3,
// v8i g4, i32 cpol); groups 2..4 are unused for <=2D tensors.
// ---------------------------------------------------------------------------
static __device__ __forceinline__ void tdm_load_2d(
    unsigned lds_addr, const bf16_t* gptr,
    unsigned tile_d0, unsigned tile_d1,
    unsigned tensor_d0, unsigned tensor_d1,
    unsigned long long stride0) {
  const unsigned long long ga = (unsigned long long)(size_t)gptr;

  v4u g0;
  g0[0] = 1u;                                    // count=1 valid user descriptor
  g0[1] = lds_addr;                              // D# bits  63:32  lds_addr
  g0[2] = (unsigned)(ga & 0xFFFFFFFFu);          // D# bits  95:64  global_addr lo
  g0[3] = (unsigned)((ga >> 32) & 0x01FFFFFFu)   // D# bits 120:96  global_addr hi
        | (2u << 30);                            // D# bits 127:126 type=2 (image)

  v8i g1;
  g1[0] = (int)((1u << 16)        // data_size = 2 bytes
              | (1u << 20)        // pad_enable
              | (4u << 22)        // pad_interval: 32 DWORDs  (= 64 bf16 row)
              | (3u << 25));      // pad_amount:   4 DWORDs  (= 8 bf16 pad)
  g1[1] = (int)((tensor_d0 & 0xFFFFu) << 16);                               // dim0 lo
  g1[2] = (int)(((tensor_d0 >> 16) & 0xFFFFu) | ((tensor_d1 & 0xFFFFu) << 16));
  g1[3] = (int)(((tensor_d1 >> 16) & 0xFFFFu) | ((tile_d0 & 0xFFFFu) << 16)); // tile_dim0
  g1[4] = (int)(tile_d1 & 0xFFFFu);                                         // tile_dim1
  g1[5] = (int)(stride0 & 0xFFFFFFFFull);                                   // dim0 stride lo
  g1[6] = (int)((stride0 >> 32) & 0xFFFFull);                               // stride hi
  g1[7] = 0;

  const v4i z4 = {0, 0, 0, 0};
  const v8i z8 = {0, 0, 0, 0, 0, 0, 0, 0};
  __builtin_amdgcn_tensor_load_to_lds(g0, g1, z4, z4, z8, 0);
}
#endif

// ---------------------------------------------------------------------------
// fp32 -> bf16 conversion, 4 elements / thread (b128 load, b64 store)
// ---------------------------------------------------------------------------
__global__ __launch_bounds__(256) void cvt_f32_bf16(const float* __restrict__ in,
                                                    bf16_t* __restrict__ out, int n) {
  const int i = (blockIdx.x * 256 + threadIdx.x) * 4;
  if (i < n) {
    const float4 f = *(const float4*)(in + i);
    v4bf o;
    o[0] = (bf16_t)f.x; o[1] = (bf16_t)f.y; o[2] = (bf16_t)f.z; o[3] = (bf16_t)f.w;
    *(v4bf*)(out + i) = o;
  }
}

// ---------------------------------------------------------------------------
// C[M,N] = A[M,K] @ W[N,K]^T   (nn.Linear, bias=false)
// Block: 256 threads (8 waves), tile 128x128, BK=64, double-buffered LDS.
// Tiles staged by the Tensor Data Mover (wave 0 posts the DMA, TENSORcnt
// tracked); 16 WMMAs per wave per BK iteration overlap the next tile's DMA.
// ---------------------------------------------------------------------------
#define LDSK 72  // padded row length (bf16): 144B stride kills the 128B bank pattern

template <typename OutT>
__global__ __launch_bounds__(256) void gemm_bf16_wmma(
    const bf16_t* __restrict__ A,   // [M,K]
    const bf16_t* __restrict__ W,   // [N,K]
    OutT* __restrict__ C,           // [M,N]
    int M, int N, int K) {
  __shared__ bf16_t As[2][128][LDSK];
  __shared__ bf16_t Bs[2][128][LDSK];

  const int tid  = threadIdx.x;
  const int lane = tid & 31;
  const int wave = tid >> 5;
  const int half = lane >> 4;    // 16-lane half of the wave32
  const int l16  = lane & 15;
  const int wm   = wave >> 1;    // 0..3 : which 32-row band
  const int wn   = wave & 1;     // 0..1 : which 64-col band
  const int m0   = blockIdx.y * 128;
  const int n0   = blockIdx.x * 128;

  v8f acc[2][4] = {};

  // stage one 128x64 tile of A and of W into LDS buffer `buf`
  auto stage = [&](int buf, int k0) {
#if USE_TDM
    if (wave == 0) {
      tdm_load_2d((unsigned)(size_t)&As[buf][0][0], A + (size_t)m0 * K + k0,
                  64u, 128u, (unsigned)K, (unsigned)M, (unsigned long long)K);
      tdm_load_2d((unsigned)(size_t)&Bs[buf][0][0], W + (size_t)n0 * K + k0,
                  64u, 128u, (unsigned)K, (unsigned)N, (unsigned long long)K);
    }
#else
#pragma unroll
    for (int it = 0; it < 4; ++it) {
      const int c   = it * 256 + tid;   // 1024 chunks of 8 bf16 per matrix
      const int row = c >> 3;
      const int col = (c & 7) * 8;
      *(v8bf*)&As[buf][row][col] = *(const v8bf*)(A + (size_t)(m0 + row) * K + k0 + col);
      *(v8bf*)&Bs[buf][row][col] = *(const v8bf*)(W + (size_t)(n0 + row) * K + k0 + col);
    }
#endif
  };

  stage(0, 0);
#if USE_TDM
  if (wave == 0) __builtin_amdgcn_s_wait_tensorcnt(0);
#endif
  __syncthreads();

  const int KT = K >> 6;   // K/64 iterations
  int buf = 0;
  for (int kt = 0; kt < KT; ++kt) {
    // post next tile's DMA into the other buffer; overlaps with WMMAs below
    if (kt + 1 < KT) stage(buf ^ 1, (kt + 1) << 6);

#pragma unroll
    for (int ks = 0; ks < 2; ++ks) {       // two 32-deep WMMA K-steps
      // A fragments (CDNA5 16-bit A 16x32 layout): lane(half,l16): M=l16,
      //   elems 0..7 -> K=8*half.., elems 8..15 -> K=16+8*half..
      v16bf af[2];
#pragma unroll
      for (int sub = 0; sub < 2; ++sub) {
        const bf16_t* p = &As[buf][wm * 32 + sub * 16 + l16][ks * 32];
        af[sub] = combine16(*(const v8bf*)(p + 8 * half),
                            *(const v8bf*)(p + 16 + 8 * half));
      }
      // B fragments (CDNA5 16-bit B 32x16 layout): lane(half,l16): N=l16,
      //   elems 0..7 -> K=8*half.., elems 8..15 -> K=16+8*half.. (same as A)
#pragma unroll
      for (int t = 0; t < 4; ++t) {
        const bf16_t* p = &Bs[buf][wn * 64 + t * 16 + l16][ks * 32 + 8 * half];
        const v16bf bfrag = combine16(*(const v8bf*)p, *(const v8bf*)(p + 16));
#pragma unroll
        for (int sub = 0; sub < 2; ++sub)
          acc[sub][t] = WMMA_BF16(af[sub], bfrag, acc[sub][t]);
      }
    }
#if USE_TDM
    if (wave == 0) __builtin_amdgcn_s_wait_tensorcnt(0);  // next tile landed
#endif
    __syncthreads();   // single barrier per BK=64 step
    buf ^= 1;
  }

  // C layout: VGPR r, lane(half,l16) -> (M = base + 8*half + r, N = t*16 + l16)
#pragma unroll
  for (int sub = 0; sub < 2; ++sub)
#pragma unroll
    for (int t = 0; t < 4; ++t)
#pragma unroll
      for (int r = 0; r < 8; ++r) {
        const int m = m0 + wm * 32 + sub * 16 + half * 8 + r;
        const int n = n0 + wn * 64 + t * 16 + l16;
        C[(size_t)m * N + n] = (OutT)acc[sub][t][r];
      }
}

// ---------------------------------------------------------------------------
// Causal flash attention over bf16 Q/K/V stored as [B*S, E] (head h = cols h*64..h*64+63).
// Grid: (S/64 query tiles, B*H). Block: 128 threads = 4 waves; wave handles 16 query rows.
// K fragments stream straight from global (contiguous in D); V is LDS-transposed.
// ---------------------------------------------------------------------------
__global__ __launch_bounds__(128) void flash_attn_wmma(
    const bf16_t* __restrict__ Q,
    const bf16_t* __restrict__ Km,
    const bf16_t* __restrict__ Vm,
    bf16_t* __restrict__ O) {
  const int S = 2048, E = 1024, D = 64;
  const float scale = 0.125f;  // 1/sqrt(64)

  const int qt   = blockIdx.x;
  const int bh   = blockIdx.y;
  const int b    = bh >> 4;
  const int h    = bh & 15;
  const int tid  = threadIdx.x;
  const int lane = tid & 31;
  const int wave = tid >> 5;
  const int half = lane >> 4;
  const int l16  = lane & 15;

  __shared__ bf16_t Pbuf[4][16][LDSK];  // per-wave probability tile (rows x kv), padded
  __shared__ bf16_t Vt[64][LDSK];       // V tile transposed: Vt[d][kv], padded

  const size_t qbase = (size_t)(b * S + qt * 64) * E + h * D;

  // Q fragments for this wave: A-matrix 16x64 -> two 16x32 k-steps, loaded from global.
  v16bf qa[2];
  {
    const bf16_t* qp = Q + qbase + (size_t)(wave * 16 + l16) * E;
#pragma unroll
    for (int ks = 0; ks < 2; ++ks)
      qa[ks] = combine16(*(const v8bf*)(qp + ks * 32 + 8 * half),
                         *(const v8bf*)(qp + ks * 32 + 16 + 8 * half));
  }

  float row_m[8], row_l[8];
  v8f o_acc[4] = {};
#pragma unroll
  for (int r = 0; r < 8; ++r) { row_m[r] = -1e30f; row_l[r] = 0.0f; }

  for (int j = 0; j <= qt; ++j) {  // causal: only kv tiles at or before the query tile
    const size_t kvbase = (size_t)(b * S + j * 64) * E + h * D;

    // --- stage V tile transposed into LDS: Vt[d][kv]
#pragma unroll
    for (int it = 0; it < 4; ++it) {
      const int c    = it * 128 + tid;   // 512 chunks of 8 bf16
      const int kv   = c >> 3;
      const int dcol = (c & 7) * 8;
      const v8bf v = *(const v8bf*)(Vm + kvbase + (size_t)kv * E + dcol);
#pragma unroll
      for (int e = 0; e < 8; ++e) Vt[dcol + e][kv] = v[e];
    }

    // --- S = Q K^T for this wave's 16 rows x 64 kv columns.
    //     K fragment (B-matrix): lane n = t*16+l16 (kv), elems 0..7 -> d = ks*32 + 8*half.., 8..15 -> d = ks*32 + 16 + 8*half..
    v8f s[4] = {};
#pragma unroll
    for (int t = 0; t < 4; ++t) {
      const bf16_t* kp = Km + kvbase + (size_t)(t * 16 + l16) * E;
#pragma unroll
      for (int ks = 0; ks < 2; ++ks) {
        const v16bf kb = combine16(*(const v8bf*)(kp + ks * 32 + 8 * half),
                                   *(const v8bf*)(kp + ks * 32 + 16 + 8 * half));
        s[t] = WMMA_BF16(qa[ks], kb, s[t]);
      }
    }

    // --- scale + causal mask
    const int qrow_base = qt * 64 + wave * 16 + half * 8;
    const int kcol_base = j * 64;
#pragma unroll
    for (int t = 0; t < 4; ++t) {
      const int kc = kcol_base + t * 16 + l16;
#pragma unroll
      for (int r = 0; r < 8; ++r) {
        const float v = s[t][r] * scale;
        s[t][r] = (kc > qrow_base + r) ? -1e30f : v;
      }
    }

    // --- online softmax, one row per (half, r); row values live in 16 lanes of the half
#pragma unroll
    for (int r = 0; r < 8; ++r) {
      float mx = -1e30f;
#pragma unroll
      for (int t = 0; t < 4; ++t) mx = fmaxf(mx, s[t][r]);
#pragma unroll
      for (int d = 1; d < 16; d <<= 1) mx = fmaxf(mx, __shfl_xor(mx, d, 32));
      const float m_new = fmaxf(row_m[r], mx);
      const float corr  = __expf(row_m[r] - m_new);
      float lsum = 0.0f;
#pragma unroll
      for (int t = 0; t < 4; ++t) {
        const float p = __expf(s[t][r] - m_new);
        s[t][r] = p;
        lsum += p;
      }
#pragma unroll
      for (int d = 1; d < 16; d <<= 1) lsum += __shfl_xor(lsum, d, 32);
      row_l[r] = row_l[r] * corr + lsum;
      row_m[r] = m_new;
#pragma unroll
      for (int t = 0; t < 4; ++t) o_acc[t][r] *= corr;
    }

    // --- round-trip P through LDS to convert C-layout -> A-layout (bf16)
#pragma unroll
    for (int t = 0; t < 4; ++t)
#pragma unroll
      for (int r = 0; r < 8; ++r)
        Pbuf[wave][half * 8 + r][t * 16 + l16] = (bf16_t)s[t][r];
    __syncthreads();

    // --- O += P V : A-frags of P, B-frags of V from Vt (contiguous in kv)
    v16bf pa[2];
#pragma unroll
    for (int ks = 0; ks < 2; ++ks) {
      const bf16_t* pp = &Pbuf[wave][l16][ks * 32];
      pa[ks] = combine16(*(const v8bf*)(pp + 8 * half),
                         *(const v8bf*)(pp + 16 + 8 * half));
    }
#pragma unroll
    for (int t = 0; t < 4; ++t) {
      const bf16_t* vp = &Vt[t * 16 + l16][0];
#pragma unroll
      for (int ks = 0; ks < 2; ++ks) {
        const v16bf vb = combine16(*(const v8bf*)(vp + ks * 32 + 8 * half),
                                   *(const v8bf*)(vp + ks * 32 + 16 + 8 * half));
        o_acc[t] = WMMA_BF16(pa[ks], vb, o_acc[t]);
      }
    }
    __syncthreads();  // protect Vt / Pbuf before next kv tile restages them
  }

  // --- normalize and store (bf16, [B*S, E] layout, head column block)
#pragma unroll
  for (int t = 0; t < 4; ++t) {
#pragma unroll
    for (int r = 0; r < 8; ++r) {
      const int m = qt * 64 + wave * 16 + half * 8 + r;
      const float val = o_acc[t][r] / row_l[r];
      O[(size_t)(b * S + m) * E + h * D + t * 16 + l16] = (bf16_t)val;
    }
  }
}

// ---------------------------------------------------------------------------
// Host-side launch
// ---------------------------------------------------------------------------
extern "C" void kernel_launch(void* const* d_in, const int* in_sizes, int n_in,
                              void* d_out, int out_size, void* d_ws, size_t ws_size,
                              hipStream_t stream) {
  const int B = 2, S = 2048, E = 1024;
  const int M = B * S;          // 4096
  const int NE = M * E;         // x / Q / K / V / attn element count (4M)
  const int NW = E * E;         // weight element count (1M)

  const float* x  = (const float*)d_in[0];
  const float* Wq = (const float*)d_in[1];
  const float* Wk = (const float*)d_in[2];
  const float* Wv = (const float*)d_in[3];
  const float* Wo = (const float*)d_in[4];

  char* ws = (char*)d_ws;
  size_t off = 0;
  bf16_t* xb   = (bf16_t*)(ws + off); off += (size_t)NE * sizeof(bf16_t);
  bf16_t* wqb  = (bf16_t*)(ws + off); off += (size_t)NW * sizeof(bf16_t);
  bf16_t* wkb  = (bf16_t*)(ws + off); off += (size_t)NW * sizeof(bf16_t);
  bf16_t* wvb  = (bf16_t*)(ws + off); off += (size_t)NW * sizeof(bf16_t);
  bf16_t* wob  = (bf16_t*)(ws + off); off += (size_t)NW * sizeof(bf16_t);
  bf16_t* Qb   = (bf16_t*)(ws + off); off += (size_t)NE * sizeof(bf16_t);
  bf16_t* Kb   = (bf16_t*)(ws + off); off += (size_t)NE * sizeof(bf16_t);
  bf16_t* Vb   = (bf16_t*)(ws + off); off += (size_t)NE * sizeof(bf16_t);
  bf16_t* Ab   = (bf16_t*)(ws + off); off += (size_t)NE * sizeof(bf16_t);
  (void)ws_size; (void)in_sizes; (void)n_in; (void)out_size;

  // 1) fp32 -> bf16 conversions (4 elems/thread)
  cvt_f32_bf16<<<NE / (256 * 4), 256, 0, stream>>>(x,  xb,  NE);
  cvt_f32_bf16<<<NW / (256 * 4), 256, 0, stream>>>(Wq, wqb, NW);
  cvt_f32_bf16<<<NW / (256 * 4), 256, 0, stream>>>(Wk, wkb, NW);
  cvt_f32_bf16<<<NW / (256 * 4), 256, 0, stream>>>(Wv, wvb, NW);
  cvt_f32_bf16<<<NW / (256 * 4), 256, 0, stream>>>(Wo, wob, NW);

  // 2) Q/K/V projections: [4096,1024] @ [1024,1024]^T (bf16 WMMA, bf16 out)
  dim3 ggrid(E / 128, M / 128);  // (8, 32)
  gemm_bf16_wmma<bf16_t><<<ggrid, 256, 0, stream>>>(xb, wqb, Qb, M, E, E);
  gemm_bf16_wmma<bf16_t><<<ggrid, 256, 0, stream>>>(xb, wkb, Kb, M, E, E);
  gemm_bf16_wmma<bf16_t><<<ggrid, 256, 0, stream>>>(xb, wvb, Vb, M, E, E);

  // 3) causal flash attention (per 64-row query tile, per (b,h))
  dim3 agrid(S / 64, B * 16);  // (32, 32)
  flash_attn_wmma<<<agrid, 128, 0, stream>>>(Qb, Kb, Vb, Ab);

  // 4) output projection -> fp32 d_out
  gemm_bf16_wmma<float><<<ggrid, 256, 0, stream>>>(Ab, wob, (float*)d_out, M, E, E);
}
